// WindowAttention_35502199668796
// MI455X (gfx1250) — matrix operand order repaired and checked by reading, not compiled, not run on hardware
//
#include <hip/hip_runtime.h>
#include <hip/hip_bf16.h>
#include <math.h>

// ---------------------------------------------------------------------------
// Swin-V2 window attention for MI455X (gfx1250, wave32, WMMA).
//   B=1024 windows, n=64 tokens, C=256, H=8 heads, d=32, NW=64 mask windows.
// Pipeline (all bf16 WMMA, fp32 accumulate):
//   1) rpb_mlp / rpb_expand : tiny MLP -> rpb[H][64][64] (16*sigmoid)
//   2) qkv_gemm             : x(65536x256) @ qkv_w^T(256x768) + bias -> q,k,v
//   3) attn                 : per (window,head): cosine attn + rpb + mask,
//                             softmax, P@V, all in LDS/registers
//   4) proj_gemm            : pre(65536x256) @ proj_w^T + proj_b -> out
// GEMM waves process 2 row-fragments (32 rows) against shared LDS weights
// (2 WMMAs per B-fragment ds_load). The k-loop is fully unrolled so the
// scheduler overlaps global_load_b128 clauses with the WMMA bursts via
// static register renaming (no rotating-register v_mov overhead).
// Workspace (floats): qkv[3*1024*8*64*32] | pre[65536*256] | rpb[8*64*64] | hbt[225*8]
// ---------------------------------------------------------------------------

typedef __bf16 bf16_t;
typedef __attribute__((ext_vector_type(16))) __bf16 v16bf;
typedef __attribute__((ext_vector_type(8)))  __bf16 v8bf;
typedef __attribute__((ext_vector_type(8)))  float  v8f;

__device__ __forceinline__ v8f wmma_bf16(v16bf a, v16bf b, v8f c) {
  // v_wmma_f32_16x16x32_bf16: (neg_a, A, neg_b, B, c_mod, C, reuse_a, reuse_b)
  return __builtin_amdgcn_wmma_f32_16x16x32_bf16(false, a, false, b,
                                                 (short)0, c, false, false);
}

// Load one 16xK A-fragment row chunk (fp32 global) and pack to bf16.
__device__ __forceinline__ v16bf load_a_frag(const float* __restrict__ row,
                                             int kb, int hl) {
  float4 l0 = *(const float4*)(row + kb + hl * 16);
  float4 l1 = *(const float4*)(row + kb + hl * 16 + 4);
  float4 l2 = *(const float4*)(row + kb + hl * 16 + 8);
  float4 l3 = *(const float4*)(row + kb + hl * 16 + 12);
  float tmp[16];
  *(float4*)&tmp[0]  = l0;
  *(float4*)&tmp[4]  = l1;
  *(float4*)&tmp[8]  = l2;
  *(float4*)&tmp[12] = l3;
  v16bf a;
#pragma unroll
  for (int i = 0; i < 16; ++i) a[i] = (bf16_t)tmp[i];
  return a;
}

// ---------------------------------------------------------------------------
// 1a) Relative-position-bias MLP: 225 table entries, 2 -> 512 (relu) -> 8
// ---------------------------------------------------------------------------
__global__ __launch_bounds__(256)
void rpb_mlp_kernel(const float* __restrict__ w1, const float* __restrict__ b1,
                    const float* __restrict__ w2, float* __restrict__ hbt) {
  int p = threadIdx.x;
  if (p >= 225) return;
  int ai = p / 15, bi = p % 15;
  float ra = (float)(ai - 7) * (8.0f / 7.0f);
  float rb = (float)(bi - 7) * (8.0f / 7.0f);
  float t0 = copysignf(log2f(fabsf(ra) + 1.0f) * (1.0f / 3.0f), ra);
  float t1 = copysignf(log2f(fabsf(rb) + 1.0f) * (1.0f / 3.0f), rb);
  float acc[8] = {0.f, 0.f, 0.f, 0.f, 0.f, 0.f, 0.f, 0.f};
  for (int j = 0; j < 512; ++j) {
    float hv = fmaxf(w1[j * 2] * t0 + w1[j * 2 + 1] * t1 + b1[j], 0.0f);
#pragma unroll
    for (int hh = 0; hh < 8; ++hh) acc[hh] += w2[hh * 512 + j] * hv;
  }
#pragma unroll
  for (int hh = 0; hh < 8; ++hh) hbt[p * 8 + hh] = acc[hh];
}

// 1b) Expand via relative-position index, 16*sigmoid -> rpb[h][i][j]
__global__ __launch_bounds__(256)
void rpb_expand_kernel(const float* __restrict__ hbt, float* __restrict__ rpb) {
  int g = blockIdx.x * 256 + threadIdx.x;         // 8*64*64 = 32768
  int h = g >> 12, rem = g & 4095, i = rem >> 6, j = rem & 63;
  int dy = (i >> 3) - (j >> 3) + 7;
  int dx = (i & 7) - (j & 7) + 7;
  float v = hbt[(dy * 15 + dx) * 8 + h];
  rpb[g] = 16.0f / (1.0f + __expf(-v));
}

// ---------------------------------------------------------------------------
// 2) QKV GEMM: M=65536, N=768, K=256. 256 thr = 8 waves; block tile 256x64,
//    wave tile 32x64 (2 A-fragments x 4 B-tiles), k-loop fully unrolled.
// ---------------------------------------------------------------------------
__global__ __launch_bounds__(256)
void qkv_gemm_kernel(const float* __restrict__ x, const float* __restrict__ qkv_w,
                     const float* __restrict__ q_bias, const float* __restrict__ v_bias,
                     float* __restrict__ qkv_out) {
  __shared__ bf16_t wt[64 * 256];                 // 64 out-cols x 256 K (32 KB)
  const int tid  = threadIdx.x;
  const int col0 = blockIdx.y * 64;
  for (int idx = tid; idx < 64 * 256; idx += 256) {
    int r = idx >> 8, k = idx & 255;
    wt[idx] = (bf16_t)qkv_w[(size_t)(col0 + r) * 256 + k];
  }
  __syncthreads();

  const int wave = tid >> 5, lane = tid & 31;
  const int l16 = lane & 15, hl = lane >> 4;
  const int m0 = blockIdx.x * 256 + wave * 32;
  const float* __restrict__ xr0 = &x[(size_t)(m0 + l16) * 256];
  const float* __restrict__ xr1 = &x[(size_t)(m0 + 16 + l16) * 256];

  v8f acc[2][4] = {{{}, {}, {}, {}}, {{}, {}, {}, {}}};
#pragma unroll
  for (int kb = 0; kb < 256; kb += 32) {
    v16bf a0 = load_a_frag(xr0, kb, hl);
    v16bf a1 = load_a_frag(xr1, kb, hl);
    v16bf bfrag[4];
#pragma unroll
    for (int t = 0; t < 4; ++t)          // batched DS loads -> one dscnt wait
      bfrag[t] = *(const v16bf*)&wt[(t * 16 + l16) * 256 + kb + hl * 16];
#pragma unroll
    for (int t = 0; t < 4; ++t) {
      acc[0][t] = wmma_bf16(a0, bfrag[t], acc[0][t]);
      acc[1][t] = wmma_bf16(a1, bfrag[t], acc[1][t]);
    }
  }
#pragma unroll
  for (int t = 0; t < 4; ++t) {
    int col = col0 + t * 16 + l16;
    int which = col >> 8, r256 = col & 255, h = r256 >> 5, dcol = r256 & 31;
    float bias = (which == 0) ? q_bias[r256] : ((which == 2) ? v_bias[r256] : 0.0f);
#pragma unroll
    for (int rf = 0; rf < 2; ++rf) {
#pragma unroll
      for (int r = 0; r < 8; ++r) {
        int m = m0 + rf * 16 + r + hl * 8;
        int b = m >> 6, tok = m & 63;
        qkv_out[((((size_t)which * 1024 + b) * 8 + h) * 64 + tok) * 32 + dcol] =
            acc[rf][t][r] + bias;
      }
    }
  }
}

// ---------------------------------------------------------------------------
// 3) Attention: one workgroup (128 thr = 4 waves) per (window, head).
// ---------------------------------------------------------------------------
__global__ __launch_bounds__(128)
void attn_kernel(const float* __restrict__ qkv, const float* __restrict__ mask,
                 const float* __restrict__ logit_scale, const float* __restrict__ rpb,
                 float* __restrict__ out_pre) {
  __shared__ bf16_t qn[64 * 32];   // normalized q
  __shared__ bf16_t kn[64 * 32];   // normalized k
  __shared__ bf16_t vT[32 * 64];   // V transposed: vT[d][k]
  __shared__ bf16_t pl[64 * 64];   // softmax probabilities

  const int bh = blockIdx.x;                 // 0 .. 8191
  const int b = bh >> 3, h = bh & 7;
  const size_t base = ((size_t)(b * 8 + h)) * 64 * 32;
  const float* __restrict__ qg = qkv + base;
  const float* __restrict__ kg = qkv + (size_t)1024 * 8 * 64 * 32 + base;
  const float* __restrict__ vg = qkv + (size_t)2 * 1024 * 8 * 64 * 32 + base;
  const int tid = threadIdx.x;

  {  // row-normalize q (threads 0..63) and k (threads 64..127) into LDS bf16
    int row = tid & 63;
    const float* src = (tid < 64) ? qg : kg;
    bf16_t* dst = (tid < 64) ? qn : kn;
    float vals[32], ss = 0.0f;
#pragma unroll
    for (int i = 0; i < 32; ++i) { float t = src[row * 32 + i]; vals[i] = t; ss += t * t; }
    float inv = 1.0f / fmaxf(sqrtf(ss), 1e-12f);
#pragma unroll
    for (int i = 0; i < 32; ++i) dst[row * 32 + i] = (bf16_t)(vals[i] * inv);
  }
  for (int idx = tid; idx < 2048; idx += 128) {      // stage V transposed
    int k = idx >> 5, dc = idx & 31;
    vT[dc * 64 + k] = (bf16_t)vg[idx];
  }
  __syncthreads();

  const int wave = tid >> 5, lane = tid & 31, l16 = lane & 15, hl = lane >> 4;

  // ---- S = qn @ kn^T  (16x64 slab per wave, K = 32, single WMMA per tile)
  v16bf a;
  {
    int row = wave * 16 + l16;
    v8bf lo = *(const v8bf*)&qn[row * 32 + hl * 16];
    v8bf hi = *(const v8bf*)&qn[row * 32 + hl * 16 + 8];
#pragma unroll
    for (int i = 0; i < 8; ++i) { a[i] = lo[i]; a[8 + i] = hi[i]; }
  }
  v16bf bk[4];
#pragma unroll
  for (int t = 0; t < 4; ++t)              // batched DS loads
    bk[t] = *(const v16bf*)&kn[(t * 16 + l16) * 32 + hl * 16];
  v8f cc[4];
  const v8f czero = {};
#pragma unroll
  for (int t = 0; t < 4; ++t) cc[t] = wmma_bf16(a, bk[t], czero);

  const float sc = __expf(fminf(logit_scale[h], 4.6051702f));   // min(ls, ln 100)
  const float* __restrict__ mrow = mask + (size_t)(b & 63) * 64 * 64;
  const float* __restrict__ rrow = rpb + (size_t)h * 64 * 64;

  float vals[4][8];
#pragma unroll
  for (int t = 0; t < 4; ++t) {
    int col = t * 16 + l16;
#pragma unroll
    for (int r = 0; r < 8; ++r) {
      int row = wave * 16 + r + hl * 8;
      vals[t][r] = cc[t][r] * sc + rrow[row * 64 + col] + mrow[row * 64 + col];
    }
  }

  // ---- softmax: each row spans its 16-lane half-group x 4 tiles (wave32)
#pragma unroll
  for (int r = 0; r < 8; ++r) {
    float mx = vals[0][r];
#pragma unroll
    for (int t = 1; t < 4; ++t) mx = fmaxf(mx, vals[t][r]);
    for (int off = 1; off < 16; off <<= 1) mx = fmaxf(mx, __shfl_xor(mx, off, 32));
    float sum = 0.0f;
#pragma unroll
    for (int t = 0; t < 4; ++t) { vals[t][r] = __expf(vals[t][r] - mx); sum += vals[t][r]; }
    for (int off = 1; off < 16; off <<= 1) sum += __shfl_xor(sum, off, 32);
    float inv = 1.0f / sum;
#pragma unroll
    for (int t = 0; t < 4; ++t) vals[t][r] *= inv;
  }

  // ---- stash P as bf16 for the P@V A-fragments
#pragma unroll
  for (int t = 0; t < 4; ++t) {
    int col = t * 16 + l16;
#pragma unroll
    for (int r = 0; r < 8; ++r) {
      int row = wave * 16 + r + hl * 8;
      pl[row * 64 + col] = (bf16_t)vals[t][r];
    }
  }
  __syncthreads();

  // ---- O = P @ V  (16x32 per wave, K = 64 as two WMMA k-steps)
  v16bf pa[2], bv[2][2];
#pragma unroll
  for (int kbi = 0; kbi < 2; ++kbi) {      // batched DS loads for all fragments
    int kb = kbi * 32;
    int row = wave * 16 + l16;
    v8bf lo = *(const v8bf*)&pl[row * 64 + kb + hl * 16];
    v8bf hi = *(const v8bf*)&pl[row * 64 + kb + hl * 16 + 8];
#pragma unroll
    for (int i = 0; i < 8; ++i) { pa[kbi][i] = lo[i]; pa[kbi][8 + i] = hi[i]; }
#pragma unroll
    for (int dt = 0; dt < 2; ++dt)
      bv[kbi][dt] = *(const v16bf*)&vT[(dt * 16 + l16) * 64 + kb + hl * 16];
  }
  v8f oacc[2] = {{}, {}};
#pragma unroll
  for (int kbi = 0; kbi < 2; ++kbi)
#pragma unroll
    for (int dt = 0; dt < 2; ++dt)
      oacc[dt] = wmma_bf16(pa[kbi], bv[kbi][dt], oacc[dt]);

#pragma unroll
  for (int dt = 0; dt < 2; ++dt) {
    int dc = dt * 16 + l16;
#pragma unroll
    for (int r = 0; r < 8; ++r) {
      int m = wave * 16 + r + hl * 8;            // token within window
      out_pre[((size_t)b * 64 + m) * 256 + h * 32 + dc] = oacc[dt][r];
    }
  }
}

// ---------------------------------------------------------------------------
// 4) Output projection: M=65536, N=256, K=256 (+ proj_b); wave tile 32x64.
// ---------------------------------------------------------------------------
__global__ __launch_bounds__(256)
void proj_gemm_kernel(const float* __restrict__ pre, const float* __restrict__ proj_w,
                      const float* __restrict__ proj_b, float* __restrict__ out) {
  __shared__ bf16_t wt[64 * 256];
  const int tid  = threadIdx.x;
  const int col0 = blockIdx.y * 64;
  for (int idx = tid; idx < 64 * 256; idx += 256) {
    int r = idx >> 8, k = idx & 255;
    wt[idx] = (bf16_t)proj_w[(size_t)(col0 + r) * 256 + k];
  }
  __syncthreads();

  const int wave = tid >> 5, lane = tid & 31;
  const int l16 = lane & 15, hl = lane >> 4;
  const int m0 = blockIdx.x * 256 + wave * 32;
  const float* __restrict__ xr0 = &pre[(size_t)(m0 + l16) * 256];
  const float* __restrict__ xr1 = &pre[(size_t)(m0 + 16 + l16) * 256];

  v8f acc[2][4] = {{{}, {}, {}, {}}, {{}, {}, {}, {}}};
#pragma unroll
  for (int kb = 0; kb < 256; kb += 32) {
    v16bf a0 = load_a_frag(xr0, kb, hl);
    v16bf a1 = load_a_frag(xr1, kb, hl);
    v16bf bfrag[4];
#pragma unroll
    for (int t = 0; t < 4; ++t)
      bfrag[t] = *(const v16bf*)&wt[(t * 16 + l16) * 256 + kb + hl * 16];
#pragma unroll
    for (int t = 0; t < 4; ++t) {
      acc[0][t] = wmma_bf16(a0, bfrag[t], acc[0][t]);
      acc[1][t] = wmma_bf16(a1, bfrag[t], acc[1][t]);
    }
  }
#pragma unroll
  for (int t = 0; t < 4; ++t) {
    int col = col0 + t * 16 + l16;
    float bias = proj_b[col];
#pragma unroll
    for (int rf = 0; rf < 2; ++rf) {
#pragma unroll
      for (int r = 0; r < 8; ++r) {
        int m = m0 + rf * 16 + r + hl * 8;
        out[(size_t)m * 256 + col] = acc[rf][t][r] + bias;
      }
    }
  }
}

// ---------------------------------------------------------------------------
extern "C" void kernel_launch(void* const* d_in, const int* in_sizes, int n_in,
                              void* d_out, int out_size, void* d_ws, size_t ws_size,
                              hipStream_t stream) {
  const float* x           = (const float*)d_in[0];
  const float* mask        = (const float*)d_in[1];
  const float* qkv_w       = (const float*)d_in[2];
  const float* q_bias      = (const float*)d_in[3];
  const float* v_bias      = (const float*)d_in[4];
  const float* logit_scale = (const float*)d_in[5];
  const float* mlp_w1      = (const float*)d_in[6];
  const float* mlp_b1      = (const float*)d_in[7];
  const float* mlp_w2      = (const float*)d_in[8];
  const float* proj_w      = (const float*)d_in[9];
  const float* proj_b      = (const float*)d_in[10];
  float* out = (float*)d_out;

  float* ws   = (float*)d_ws;
  float* qkv  = ws;                                   // 3*1024*8*64*32 = 50331648
  float* pre  = qkv + (size_t)50331648;               // 65536*256      = 16777216
  float* rpb  = pre + (size_t)16777216;               // 8*64*64        = 32768
  float* hbt  = rpb + (size_t)32768;                  // 225*8

  rpb_mlp_kernel<<<1, 256, 0, stream>>>(mlp_w1, mlp_b1, mlp_w2, hbt);
  rpb_expand_kernel<<<128, 256, 0, stream>>>(hbt, rpb);
  qkv_gemm_kernel<<<dim3(256, 12), 256, 0, stream>>>(x, qkv_w, q_bias, v_bias, qkv);
  attn_kernel<<<8192, 128, 0, stream>>>(qkv, mask, logit_scale, rpb, pre);
  proj_gemm_kernel<<<dim3(256, 4), 256, 0, stream>>>(pre, proj_w, proj_b, out);
}
